// ResidualVectorQuantizer_19928648253696
// MI455X (gfx1250) — matrix operand order repaired and checked by reading, not compiled, not run on hardware
//
#include <hip/hip_runtime.h>
#include <hip/hip_bf16.h>

typedef _Float16 half16 __attribute__((ext_vector_type(16)));
typedef float    float8 __attribute__((ext_vector_type(8)));

#define B_  8
#define D_  128
#define T_  4096
#define Q_  30
#define C_  1024

#define ROWS  128       // rows (B*T slices) per block
#define RSTR  130       // residual LDS row stride in floats (pad vs 64 banks)
#define CHUNK 128       // codes staged per LDS chunk
#define CSTR  136       // chunk LDS row stride in halves (pad vs 64 banks)

__global__ __launch_bounds__(256)
void rvq_kernel(const float* __restrict__ x, const float* __restrict__ cb,
                float* __restrict__ out, float* __restrict__ loss)
{
    __shared__ float    res[ROWS * RSTR];       // f32 residual, resident for all Q steps
    __shared__ _Float16 cbs[CHUNK * CSTR];      // f16 staged codebook chunk
    __shared__ float    cnorm[CHUNK];           // ||c||^2 for staged chunk
    __shared__ float    red[256];               // loss reduction

    const int tid  = threadIdx.x;
    const int lane = tid & 31;
    const int wave = tid >> 5;
    const int l15  = lane & 15;
    const int hi   = lane >> 4;                 // half-wave id
    const int blk  = blockIdx.x;
    const int b    = blk >> 5;                  // 32 blocks per batch (4096/128)
    const int t0   = (blk & 31) * ROWS;

    const float* xblk = x + (size_t)b * D_ * T_ + t0;

    // ---- init residual: transpose x[b, d, t0+tl] -> res[tl][d] ----
    {
        const int tl = tid & 127;
        const int db = tid >> 7;
        #pragma unroll 4
        for (int i = 0; i < 64; ++i) {
            int d = i * 2 + db;
            res[tl * RSTR + d] = xblk[(size_t)d * T_ + tl];
        }
    }
    __syncthreads();

    const int rowBase = wave * 16;              // this wave's 16-row tile
    float lacc = 0.0f;

    #pragma unroll 1
    for (int q = 0; q < Q_; ++q) {
        const float* cbq = cb + (size_t)q * C_ * D_;

        // ---- build f16 A-fragments for this wave's rows (contiguous k order: k = 16*hi + e, same as B) ----
        half16 afrag[4];
        {
            const float* rrow = &res[(rowBase + l15) * RSTR];
            #pragma unroll
            for (int kc = 0; kc < 4; ++kc) {
                const int k0 = kc * 32 + hi * 16;       // K = {0..7} or {16..23}
                const int k1 = k0 + 8;                  // K = {8..15} or {24..31}
                #pragma unroll
                for (int e = 0; e < 8; ++e) {
                    afrag[kc][e]     = (_Float16)rrow[k0 + e];
                    afrag[kc][e + 8] = (_Float16)rrow[k1 + e];
                }
            }
        }

        float best[8];
        int   bidx[8];
        #pragma unroll
        for (int j = 0; j < 8; ++j) { best[j] = 3.0e38f; bidx[j] = 0x7fffffff; }

        #pragma unroll 1
        for (int ch = 0; ch < C_ / CHUNK; ++ch) {
            __syncthreads();   // previous chunk fully consumed by all waves

            // ---- stage 128 codes as f16 (coalesced float4 reads) ----
            {
                const float4* src = (const float4*)(cbq + (size_t)ch * CHUNK * D_);
                #pragma unroll
                for (int i = 0; i < 16; ++i) {
                    float4 v = src[i * 256 + tid];
                    int e4   = i * 1024 + tid * 4;      // flat float index in chunk
                    int code = e4 >> 7;
                    int dim  = e4 & 127;
                    _Float16* dst = &cbs[code * CSTR + dim];
                    dst[0] = (_Float16)v.x; dst[1] = (_Float16)v.y;
                    dst[2] = (_Float16)v.z; dst[3] = (_Float16)v.w;
                }
            }
            __syncthreads();

            // ---- chunk norms from staged f16 (2 threads per code) ----
            {
                const int code = tid >> 1;
                const _Float16* row = &cbs[code * CSTR + (tid & 1) * 64];
                float s = 0.0f;
                #pragma unroll 8
                for (int d0 = 0; d0 < 64; ++d0) { float f = (float)row[d0]; s += f * f; }
                s += __shfl_xor(s, 1, 32);
                if ((tid & 1) == 0) cnorm[code] = s;
            }
            __syncthreads();

            // ---- 8 code tiles of 16: scores via WMMA, running argmin ----
            // NOTE: candidate index n increases strictly over the scan, so a
            // strict '<' keeps the FIRST minimum (argmin semantics) with no
            // tie-break compares in the hot loop.
            #pragma unroll 2
            for (int ct = 0; ct < CHUNK / 16; ++ct) {
                float8 acc = {};
                #pragma unroll
                for (int kc = 0; kc < 4; ++kc) {
                    // dense 16-bit B layout: lane&15 = column, half-wave picks K 0..15 / 16..31
                    half16 bfrag;
                    const _Float16* brow = &cbs[(ct * 16 + l15) * CSTR + kc * 32 + hi * 16];
                    #pragma unroll
                    for (int e = 0; e < 16; ++e) bfrag[e] = brow[e];
                    acc = __builtin_amdgcn_wmma_f32_16x16x32_f16(
                              false, afrag[kc], false, bfrag,
                              (short)0, acc, false, false);
                }
                const int   n  = ch * CHUNK + ct * 16 + l15;
                const float nv = cnorm[ct * 16 + l15];
                #pragma unroll
                for (int j = 0; j < 8; ++j) {
                    float sc = nv - 2.0f * acc[j];      // ||c||^2 - 2 r.c
                    if (sc < best[j]) { best[j] = sc; bidx[j] = n; }
                }
            }
        }

        // ---- per-row argmin across the 16 lanes of each half-wave ----
        // (tie-break on lower index here, where order is not monotone)
        #pragma unroll
        for (int j = 0; j < 8; ++j) {
            #pragma unroll
            for (int m = 1; m < 16; m <<= 1) {
                float ob = __shfl_xor(best[j], m, 16);
                int   oi = __shfl_xor(bidx[j], m, 16);
                if (ob < best[j] || (ob == best[j] && oi < bidx[j])) {
                    best[j] = ob; bidx[j] = oi;
                }
            }
        }

        // ---- gather exact f32 codeword, update residual, accumulate loss ----
        // half-wave `hi` owns rows rowBase + hi*8 + j; each lane covers 8 dims
        {
            const int dl = l15 * 8;
            #pragma unroll
            for (int j = 0; j < 8; ++j) {
                const int row = rowBase + hi * 8 + j;
                const float* qv = cbq + (size_t)bidx[j] * D_ + dl;
                float* rp = &res[row * RSTR + dl];
                #pragma unroll
                for (int e = 0; e < 8; ++e) {
                    float nr = rp[e] - qv[e];
                    rp[e] = nr;
                    lacc += nr * nr;                    // loss_q = mean(new_res^2)
                }
            }
        }
        // residual rows are private to this wave; chunk barrier protects cbs reuse
    }

    __syncthreads();

    // ---- quantized = x - final_residual (sum of quants telescopes) ----
    {
        const int tl = tid & 127;
        const int db = tid >> 7;
        float* oblk = out + (size_t)b * D_ * T_ + t0;
        #pragma unroll 4
        for (int i = 0; i < 64; ++i) {
            int d = i * 2 + db;
            oblk[(size_t)d * T_ + tl] = xblk[(size_t)d * T_ + tl] - res[tl * RSTR + d];
        }
    }

    // ---- commit loss: block reduce + atomic ----
    red[tid] = lacc;
    __syncthreads();
    for (int s = 128; s > 0; s >>= 1) {
        if (tid < s) red[tid] += red[tid + s];
        __syncthreads();
    }
    if (tid == 0)
        atomicAdd(loss, red[0] * (1.0f / ((float)B_ * (float)T_ * (float)D_)));
}

extern "C" void kernel_launch(void* const* d_in, const int* in_sizes, int n_in,
                              void* d_out, int out_size, void* d_ws, size_t ws_size,
                              hipStream_t stream) {
    (void)in_sizes; (void)n_in; (void)d_ws; (void)ws_size; (void)out_size;
    const float* x  = (const float*)d_in[0];     // [B, D, T]
    const float* cb = (const float*)d_in[1];     // [Q, C, D]
    float* out  = (float*)d_out;                 // [B, D, T] quantized, then scalar loss
    float* loss = out + (size_t)B_ * D_ * T_;

    hipMemsetAsync(loss, 0, sizeof(float), stream);   // capture-legal
    rvq_kernel<<<dim3((B_ * T_) / ROWS), dim3(256), 0, stream>>>(x, cb, out, loss);
}
